// SelfAttention_2087354106673
// MI455X (gfx1250) — matrix operand order repaired and checked by reading, compiled, not run on hardware
//
#include <hip/hip_runtime.h>
#include <hip/hip_bf16.h>

// CDNA5 / gfx1250: wave32, WMMA 16x16x32 bf16 (f32 accumulate).
typedef __attribute__((ext_vector_type(16))) __bf16 v16bf;
typedef __attribute__((ext_vector_type(8)))  float  v8f;
typedef __attribute__((ext_vector_type(4)))  int    v4i;

#define D_MODEL   512
#define NUM_HEADS 8
#define HEAD_DIM  64
#define BATCH     4
#define SEQ       2048
#define M_TOTAL   (BATCH * SEQ)   // 8192 rows

#define AS_GLOBAL __attribute__((address_space(1)))
#define AS_LDS    __attribute__((address_space(3)))

#ifndef __has_builtin
#define __has_builtin(x) 0
#endif
// gfx1250 async global->LDS DMA path (ASYNCcnt). Gated: fall back to
// vectorized b128 loads if this toolchain doesn't declare the builtins.
#if __has_builtin(__builtin_amdgcn_global_load_async_to_lds_b128) && \
    __has_builtin(__builtin_amdgcn_s_wait_asynccnt)
#define HAVE_ASYNC_LDS 1
#else
#define HAVE_ASYNC_LDS 0
#endif

__device__ __forceinline__ void async_or_copy_b128(const __bf16* g, __bf16* l) {
#if HAVE_ASYNC_LDS
    // Signature (from clang diagnostic): (int4 AS1*, int4 AS3*, imm offset, imm cpol)
    __builtin_amdgcn_global_load_async_to_lds_b128(
        (AS_GLOBAL v4i*)g, (AS_LDS v4i*)l, 0, 0);
#else
    *(uint4*)l = *(const uint4*)g;
#endif
}
__device__ __forceinline__ void async_wait_all() {
#if HAVE_ASYNC_LDS
    __builtin_amdgcn_s_wait_asynccnt(0);
#endif
}

// ---------------------------------------------------------------------------
// Kernel 0: convert weight fp32 [N][K] -> bf16 transposed [K][N]
// ---------------------------------------------------------------------------
__global__ void wt_convert(const float* __restrict__ W, __bf16* __restrict__ Wt) {
    int idx = blockIdx.x * blockDim.x + threadIdx.x;   // over 512*512
    int n = idx >> 9, k = idx & 511;
    Wt[(size_t)k * D_MODEL + n] = (__bf16)W[(size_t)n * D_MODEL + k];
}

// ---------------------------------------------------------------------------
// Tiled WMMA GEMM: out = X[M x 512] @ W^T + bias, W given pre-transposed bf16.
// Block tile 128x64, K-step 32. 8 waves in 4x2 grid; each wave: 32x32 via
// 2x2 v_wmma_f32_16x16x32_bf16. B tile stored n-major in LDS so B-fragment
// reads are contiguous ds_load_b128.
// ---------------------------------------------------------------------------
template <typename XT, bool HEAD_OUT>
__global__ __launch_bounds__(256)
void gemm512_wmma(const XT* __restrict__ X, const __bf16* __restrict__ Wt,
                  const float* __restrict__ bias, void* __restrict__ outp) {
    constexpr int BM = 128, BN = 64, BK = 32, PA = 8, PB = 8;
    __shared__ __bf16 As[BM][BK + PA];        // row-major (m, k)
    __shared__ __bf16 Bst[BN][BK + PB];       // transposed: (n, k)

    const int tid  = threadIdx.x;
    const int lane = tid & 31;
    const int wave = tid >> 5;      // 0..7
    const int wy   = wave >> 1;     // 0..3 (row group of 32)
    const int wx   = wave & 1;      // 0..1 (col group of 32)
    const int hl   = lane >> 4;     // half-wave select
    const int l15  = lane & 15;
    const int m0   = blockIdx.x * BM;
    const int n0   = blockIdx.y * BN;

    v8f acc[2][2];
#pragma unroll
    for (int i = 0; i < 2; ++i)
#pragma unroll
        for (int j = 0; j < 2; ++j)
#pragma unroll
            for (int e = 0; e < 8; ++e) acc[i][j][e] = 0.0f;

    for (int k0 = 0; k0 < D_MODEL; k0 += BK) {
        __syncthreads();
        // ---- A tile: 128x32, b128 global loads (convert fp32->bf16 if needed)
        if constexpr (sizeof(XT) == 4) {
#pragma unroll
            for (int i = 0; i < 4; ++i) {           // 1024 float4 chunks
                int c = tid + i * 256;
                int r = c >> 3, col = (c & 7) * 4;
                float4 f = *(const float4*)&X[(size_t)(m0 + r) * D_MODEL + k0 + col];
                union { uint2 u; __bf16 h[4]; } t;
                t.h[0] = (__bf16)f.x; t.h[1] = (__bf16)f.y;
                t.h[2] = (__bf16)f.z; t.h[3] = (__bf16)f.w;
                *(uint2*)&As[r][col] = t.u;
            }
        } else {
#pragma unroll
            for (int i = 0; i < 2; ++i) {           // 512 b128 chunks
                int c = tid + i * 256;
                int r = c >> 2, col = (c & 3) * 8;
                async_or_copy_b128(&X[(size_t)(m0 + r) * D_MODEL + k0 + col], &As[r][col]);
            }
        }
        // ---- B tile: 32x64 read coalesced, stored transposed (n, k)
        {
            int c = tid;                            // exactly 256 b128 chunks
            int k = c >> 3, n = (c & 7) * 8;
            uint4 w8 = *(const uint4*)&Wt[(size_t)(k0 + k) * D_MODEL + n0 + n];
            union { uint4 u; __bf16 h[8]; } t; t.u = w8;
#pragma unroll
            for (int j = 0; j < 8; ++j) Bst[n + j][k] = t.h[j];
        }
        if (k0 + BK < D_MODEL) {   // hint next A tile -> global_prefetch_b8
            __builtin_prefetch(&X[(size_t)(m0 + (tid >> 1)) * D_MODEL + k0 + BK], 0, 0);
        }
        if constexpr (sizeof(XT) != 4) async_wait_all();
        __syncthreads();

        // A fragments (lanes0-15 K{0-15}, lanes16-31 K{16-31})
        v16bf a[2], b[2];
#pragma unroll
        for (int i = 0; i < 2; ++i) {
            int mr = wy * 32 + i * 16 + l15;
            int kb = hl * 16;
#pragma unroll
            for (int e = 0; e < 8; ++e) a[i][e]     = As[mr][kb + e];
#pragma unroll
            for (int e = 0; e < 8; ++e) a[i][8 + e] = As[mr][kb + 8 + e];
        }
        // B fragments: contiguous along k in Bst -> 2x ds_load_b128 each
#pragma unroll
        for (int j = 0; j < 2; ++j) {
            int nc = wx * 32 + j * 16 + l15;
            int kb = hl * 16;
#pragma unroll
            for (int e = 0; e < 16; ++e) b[j][e] = Bst[nc][kb + e];
        }
#pragma unroll
        for (int i = 0; i < 2; ++i)
#pragma unroll
            for (int j = 0; j < 2; ++j)
                acc[i][j] = __builtin_amdgcn_wmma_f32_16x16x32_bf16(
                    false, a[i], false, b[j], (short)0, acc[i][j], false, false);
    }

    // Epilogue. C layout: row m = r + 8*hl, col n = l15.
#pragma unroll
    for (int i = 0; i < 2; ++i) {
#pragma unroll
        for (int j = 0; j < 2; ++j) {
#pragma unroll
            for (int r = 0; r < 8; ++r) {
                int g = m0 + wy * 32 + i * 16 + r + 8 * hl;   // global row
                int c = n0 + wx * 32 + j * 16 + l15;          // global col
                float v = acc[i][j][r] + bias[c];
                if (HEAD_OUT) {
                    int b_ = g >> 11, s = g & (SEQ - 1);
                    int h = c >> 6, d = c & (HEAD_DIM - 1);
                    ((__bf16*)outp)[((size_t)(b_ * NUM_HEADS + h) * SEQ + s) * HEAD_DIM + d] =
                        (__bf16)v;
                } else {
                    ((float*)outp)[(size_t)g * D_MODEL + c] = v;
                }
            }
        }
    }
}

// ---------------------------------------------------------------------------
// Flash attention: one block = 128 query rows of one (b,h). 8 waves, each owns
// 16 query rows. Online softmax over 64-key blocks. V kept d-major in LDS so
// the P@V B-fragments read contiguously.
// ---------------------------------------------------------------------------
__global__ __launch_bounds__(256)
void flash_attn(const __bf16* __restrict__ Qh, const __bf16* __restrict__ Kh,
                const __bf16* __restrict__ Vh, __bf16* __restrict__ Ctx) {
    constexpr int QT = 128, KT = 64, LDW = HEAD_DIM + 8;   // 72: keeps 16B align
    __shared__ __bf16 Qs[QT][LDW];        // (q, d)
    __shared__ __bf16 Ks[KT][LDW];        // (key, d)
    __shared__ __bf16 Vt[HEAD_DIM][LDW];  // transposed: (d, key)
    __shared__ __bf16 Ps[8][16][LDW];     // per-wave P tile (C->A layout bounce)

    const int tid  = threadIdx.x;
    const int lane = tid & 31;
    const int wave = tid >> 5;
    const int hl   = lane >> 4;
    const int l15  = lane & 15;
    const int q0   = blockIdx.x * QT;
    const int bh   = blockIdx.y;                  // b*8 + h
    const size_t base = (size_t)bh * SEQ * HEAD_DIM;

    // Q tile (128x64): 1024 b128 chunks, 4/thread (async DMA if available).
#pragma unroll
    for (int i = 0; i < 4; ++i) {
        int c = tid + i * 256;
        int r = c >> 3, col = (c & 7) * 8;
        async_or_copy_b128(&Qh[base + (size_t)(q0 + r) * HEAD_DIM + col], &Qs[r][col]);
    }
    async_wait_all();
    __syncthreads();

    // Preload this wave's Q fragments (two 16x32 A frags), reused all blocks.
    v16bf qa[2];
#pragma unroll
    for (int ks = 0; ks < 2; ++ks) {
        int mr = wave * 16 + l15;
        int kb = ks * 32 + hl * 16;
#pragma unroll
        for (int e = 0; e < 8; ++e) qa[ks][e]     = Qs[mr][kb + e];
#pragma unroll
        for (int e = 0; e < 8; ++e) qa[ks][8 + e] = Qs[mr][kb + 8 + e];
    }

    float mrow[8], lrow[8];
#pragma unroll
    for (int r = 0; r < 8; ++r) { mrow[r] = -1e30f; lrow[r] = 0.0f; }
    v8f o[4];
#pragma unroll
    for (int j = 0; j < 4; ++j)
#pragma unroll
        for (int e = 0; e < 8; ++e) o[j][e] = 0.0f;

    const float scale = 0.125f;   // 1/sqrt(64)

    for (int kb0 = 0; kb0 < SEQ; kb0 += KT) {
        __syncthreads();   // all waves done reading previous K/V tiles
        // K tile row-major: 512 b128 chunks, 2/thread (async DMA if available).
#pragma unroll
        for (int i = 0; i < 2; ++i) {
            int c = tid + i * 256;
            int r = c >> 3, col = (c & 7) * 8;
            async_or_copy_b128(&Kh[base + (size_t)(kb0 + r) * HEAD_DIM + col], &Ks[r][col]);
        }
        // V tile: b128 global read, transposed scalar stores -> Vt[d][key].
#pragma unroll
        for (int i = 0; i < 2; ++i) {
            int c = tid + i * 256;
            int r = c >> 3, col = (c & 7) * 8;
            uint4 vv = *(const uint4*)&Vh[base + (size_t)(kb0 + r) * HEAD_DIM + col];
            union { uint4 u; __bf16 h[8]; } t; t.u = vv;
#pragma unroll
            for (int j = 0; j < 8; ++j) Vt[col + j][r] = t.h[j];
        }
        async_wait_all();
        __syncthreads();

        // Scores S = Q(16x64) @ K^T: 4 N-tiles x 2 K-steps of WMMA.
        v8f s[4];
#pragma unroll
        for (int j = 0; j < 4; ++j)
#pragma unroll
            for (int e = 0; e < 8; ++e) s[j][e] = 0.0f;
#pragma unroll
        for (int ks = 0; ks < 2; ++ks) {
#pragma unroll
            for (int j = 0; j < 4; ++j) {
                v16bf bfr;                         // B[d][key] = K[key][d]
                int n  = j * 16 + l15;             // key column (lane-major)
                int kb = ks * 32 + hl * 16;        // d index base (contiguous)
#pragma unroll
                for (int e = 0; e < 16; ++e) bfr[e] = Ks[n][kb + e];
                s[j] = __builtin_amdgcn_wmma_f32_16x16x32_bf16(
                    false, qa[ks], false, bfr, (short)0, s[j], false, false);
            }
        }

        // Online softmax. Row m = r + 8*hl lives across the 16 lanes of a half.
        float mnew[8], alpha[8];
#pragma unroll
        for (int r = 0; r < 8; ++r) {
            float v = -1e30f;
#pragma unroll
            for (int j = 0; j < 4; ++j) v = fmaxf(v, s[j][r]);
            v *= scale;
#pragma unroll
            for (int msk = 1; msk <= 8; msk <<= 1) v = fmaxf(v, __shfl_xor(v, msk, 32));
            mnew[r]  = fmaxf(mrow[r], v);
            alpha[r] = __expf(mrow[r] - mnew[r]);
            mrow[r]  = mnew[r];
        }
#pragma unroll
        for (int r = 0; r < 8; ++r) {
            float rsum = 0.0f;
#pragma unroll
            for (int j = 0; j < 4; ++j) {
                float p = __expf(s[j][r] * scale - mnew[r]);
                rsum += p;
                Ps[wave][r + 8 * hl][j * 16 + l15] = (__bf16)p;   // C-layout -> LDS
            }
#pragma unroll
            for (int msk = 1; msk <= 8; msk <<= 1) rsum += __shfl_xor(rsum, msk, 32);
            lrow[r] = lrow[r] * alpha[r] + rsum;
#pragma unroll
            for (int j = 0; j < 4; ++j) o[j][r] *= alpha[r];
        }

        // O += P(16x64) @ V. A frags from Ps (same-wave LDS, in-order),
        // B frags contiguous from transposed Vt.
#pragma unroll
        for (int ks = 0; ks < 2; ++ks) {
            v16bf pa;
            int mr = l15;
            int kb = ks * 32 + hl * 16;
#pragma unroll
            for (int e = 0; e < 8; ++e) pa[e]     = Ps[wave][mr][kb + e];
#pragma unroll
            for (int e = 0; e < 8; ++e) pa[8 + e] = Ps[wave][mr][kb + 8 + e];
#pragma unroll
            for (int j = 0; j < 4; ++j) {
                v16bf bfr;                          // B[l][d] = Vt[d][l]
                int n   = j * 16 + l15;             // output dim column
                int kb2 = ks * 32 + hl * 16;        // key index base (contiguous)
#pragma unroll
                for (int e = 0; e < 16; ++e) bfr[e] = Vt[n][kb2 + e];
                o[j] = __builtin_amdgcn_wmma_f32_16x16x32_bf16(
                    false, pa, false, bfr, (short)0, o[j], false, false);
            }
        }
    }

    // Normalize and store context in [b, s, h, d] (= flat [8192, 512]).
    const int b_ = bh >> 3, h = bh & 7;
#pragma unroll
    for (int r = 0; r < 8; ++r) {
        float inv = 1.0f / lrow[r];
        int qrow = q0 + wave * 16 + r + 8 * hl;
#pragma unroll
        for (int j = 0; j < 4; ++j) {
            int d = j * 16 + l15;
            Ctx[((size_t)(b_ * SEQ + qrow) * NUM_HEADS + h) * HEAD_DIM + d] =
                (__bf16)(o[j][r] * inv);
        }
    }
}

// ---------------------------------------------------------------------------
// Launch: weights->bf16^T, three projection GEMMs, flash attention, out GEMM.
// Workspace: 4 x 512KB transposed weights + 4 x 8MB bf16 tensors (~34 MB).
// ---------------------------------------------------------------------------
extern "C" void kernel_launch(void* const* d_in, const int* in_sizes, int n_in,
                              void* d_out, int out_size, void* d_ws, size_t ws_size,
                              hipStream_t stream) {
    const float* q    = (const float*)d_in[0];
    const float* k    = (const float*)d_in[1];
    const float* v    = (const float*)d_in[2];
    const float* WQ_w = (const float*)d_in[3];
    const float* WQ_b = (const float*)d_in[4];
    const float* WK_w = (const float*)d_in[5];
    const float* WK_b = (const float*)d_in[6];
    const float* WV_w = (const float*)d_in[7];
    const float* WV_b = (const float*)d_in[8];
    const float* FO_w = (const float*)d_in[9];
    const float* FO_b = (const float*)d_in[10];
    float* out = (float*)d_out;

    char* ws = (char*)d_ws;
    const size_t WBYTES = (size_t)D_MODEL * D_MODEL * sizeof(__bf16);
    const size_t QKVB   = (size_t)BATCH * NUM_HEADS * SEQ * HEAD_DIM * sizeof(__bf16);
    __bf16* WQt = (__bf16*)ws;              ws += WBYTES;
    __bf16* WKt = (__bf16*)ws;              ws += WBYTES;
    __bf16* WVt = (__bf16*)ws;              ws += WBYTES;
    __bf16* FOt = (__bf16*)ws;              ws += WBYTES;
    __bf16* Qh  = (__bf16*)ws;              ws += QKVB;
    __bf16* Kh  = (__bf16*)ws;              ws += QKVB;
    __bf16* Vh  = (__bf16*)ws;              ws += QKVB;
    __bf16* Ctx = (__bf16*)ws;              ws += QKVB;

    wt_convert<<<(D_MODEL * D_MODEL) / 256, 256, 0, stream>>>(WQ_w, WQt);
    wt_convert<<<(D_MODEL * D_MODEL) / 256, 256, 0, stream>>>(WK_w, WKt);
    wt_convert<<<(D_MODEL * D_MODEL) / 256, 256, 0, stream>>>(WV_w, WVt);
    wt_convert<<<(D_MODEL * D_MODEL) / 256, 256, 0, stream>>>(FO_w, FOt);

    dim3 gg(M_TOTAL / 128, D_MODEL / 64);   // (64, 8)
    gemm512_wmma<float, true><<<gg, 256, 0, stream>>>(q, WQt, WQ_b, (void*)Qh);
    gemm512_wmma<float, true><<<gg, 256, 0, stream>>>(k, WKt, WK_b, (void*)Kh);
    gemm512_wmma<float, true><<<gg, 256, 0, stream>>>(v, WVt, WV_b, (void*)Vh);

    flash_attn<<<dim3(SEQ / 128, BATCH * NUM_HEADS), 256, 0, stream>>>(Qh, Kh, Vh, Ctx);

    gemm512_wmma<__bf16, false><<<gg, 256, 0, stream>>>(Ctx, FOt, FO_b, (void*)out);
}
